// AttackModule_28174985462502
// MI455X (gfx1250) — matrix operand order repaired and checked by reading, not compiled, not run on hardware
//
#include <hip/hip_runtime.h>

typedef __attribute__((ext_vector_type(16))) _Float16 v16h;
typedef __attribute__((ext_vector_type(4)))  _Float16 v4h;
typedef __attribute__((ext_vector_type(8)))  float    v8f;

#define NEG_SLOPE 0.01f
#define VERY_LARGE_NUMBER 1e10f
#define D_FEAT 64
#define H_DIM 128

// ---------------------------------------------------------------------------
// Kernel 0: transpose + convert W1 (128x128 f32, row-major [k][col]) into
// W1T (f16, column-major: W1T[col*128 + k]). 32 KB -> stays resident in L0/L2.
// ---------------------------------------------------------------------------
__global__ __launch_bounds__(256)
void w1_transpose_f16(const float* __restrict__ W1, _Float16* __restrict__ W1T) {
    const int i = blockIdx.x * blockDim.x + threadIdx.x;   // dest index
    if (i < 2 * D_FEAT * H_DIM) {
        const int col = i >> 7;          // 0..127
        const int k   = i & 127;         // 0..127
        W1T[i] = (_Float16)W1[k * H_DIM + col];
    }
}

// ---------------------------------------------------------------------------
// Kernel 1: precompute A = feat @ W1[0:64,:] + b1  and  P = feat @ W1[64:128,:]
// with f16 WMMA (f32 accumulate). Block = 256 threads = 8 waves; one 16-row
// M-tile per wave. B fragments come as contiguous 32-byte loads from W1T.
// ---------------------------------------------------------------------------
__global__ __launch_bounds__(256)
void attack_precompute_wmma(const float* __restrict__ feat,
                            const _Float16* __restrict__ W1T,
                            const float* __restrict__ b1,
                            _Float16* __restrict__ Ah,
                            _Float16* __restrict__ Ph,
                            int N, int numMTiles) {
    const int tid  = threadIdx.x;
    const int wave = tid >> 5;
    const int lane = tid & 31;
    const int mtile = blockIdx.x * 8 + wave;
    if (mtile >= numMTiles) return;

    // ---- A-fragments (feat rows, K = 0..63 as two 16x32 f16 frags) ----
    // Layout, 16-bit A 16x32 (contiguous, matches B): lane<16 -> row M=lane, K in {0..15};
    //                             lane>=16 -> row M=lane-16, K in {16..31}.
    int row = mtile * 16 + (lane & 15);
    if (row >= N) row = N - 1;
    const float4* fr4 = reinterpret_cast<const float4*>(feat + (size_t)row * D_FEAT);
    const int kq = (lane < 16) ? 0 : 4;              // kbase/4 (kbase = 0 or 16)

    float4 q0 = fr4[kq + 0],  q1 = fr4[kq + 1];      // K = kbase    .. kbase+7
    float4 q2 = fr4[kq + 2],  q3 = fr4[kq + 3];      // K = kbase+8  .. kbase+15
    float4 q4 = fr4[kq + 8],  q5 = fr4[kq + 9];      // K = 32+kbase .. 32+kbase+7
    float4 q6 = fr4[kq + 10], q7 = fr4[kq + 11];     // K = 40+kbase .. 40+kbase+7

#if __has_builtin(__builtin_amdgcn_cvt_pkrtz)
    using pkh2 = decltype(__builtin_amdgcn_cvt_pkrtz(0.0f, 0.0f));
    union Frag { v16h v; pkh2 p[8]; };
    Frag ua0, ua1;
    ua0.p[0] = __builtin_amdgcn_cvt_pkrtz(q0.x, q0.y);
    ua0.p[1] = __builtin_amdgcn_cvt_pkrtz(q0.z, q0.w);
    ua0.p[2] = __builtin_amdgcn_cvt_pkrtz(q1.x, q1.y);
    ua0.p[3] = __builtin_amdgcn_cvt_pkrtz(q1.z, q1.w);
    ua0.p[4] = __builtin_amdgcn_cvt_pkrtz(q2.x, q2.y);
    ua0.p[5] = __builtin_amdgcn_cvt_pkrtz(q2.z, q2.w);
    ua0.p[6] = __builtin_amdgcn_cvt_pkrtz(q3.x, q3.y);
    ua0.p[7] = __builtin_amdgcn_cvt_pkrtz(q3.z, q3.w);
    ua1.p[0] = __builtin_amdgcn_cvt_pkrtz(q4.x, q4.y);
    ua1.p[1] = __builtin_amdgcn_cvt_pkrtz(q4.z, q4.w);
    ua1.p[2] = __builtin_amdgcn_cvt_pkrtz(q5.x, q5.y);
    ua1.p[3] = __builtin_amdgcn_cvt_pkrtz(q5.z, q5.w);
    ua1.p[4] = __builtin_amdgcn_cvt_pkrtz(q6.x, q6.y);
    ua1.p[5] = __builtin_amdgcn_cvt_pkrtz(q6.z, q6.w);
    ua1.p[6] = __builtin_amdgcn_cvt_pkrtz(q7.x, q7.y);
    ua1.p[7] = __builtin_amdgcn_cvt_pkrtz(q7.z, q7.w);
    v16h a0 = ua0.v, a1 = ua1.v;
#else
    v16h a0, a1;
    a0[0]=(_Float16)q0.x; a0[1]=(_Float16)q0.y; a0[2]=(_Float16)q0.z; a0[3]=(_Float16)q0.w;
    a0[4]=(_Float16)q1.x; a0[5]=(_Float16)q1.y; a0[6]=(_Float16)q1.z; a0[7]=(_Float16)q1.w;
    a0[8]=(_Float16)q2.x; a0[9]=(_Float16)q2.y; a0[10]=(_Float16)q2.z; a0[11]=(_Float16)q2.w;
    a0[12]=(_Float16)q3.x; a0[13]=(_Float16)q3.y; a0[14]=(_Float16)q3.z; a0[15]=(_Float16)q3.w;
    a1[0]=(_Float16)q4.x; a1[1]=(_Float16)q4.y; a1[2]=(_Float16)q4.z; a1[3]=(_Float16)q4.w;
    a1[4]=(_Float16)q5.x; a1[5]=(_Float16)q5.y; a1[6]=(_Float16)q5.z; a1[7]=(_Float16)q5.w;
    a1[8]=(_Float16)q6.x; a1[9]=(_Float16)q6.y; a1[10]=(_Float16)q6.z; a1[11]=(_Float16)q6.w;
    a1[12]=(_Float16)q7.x; a1[13]=(_Float16)q7.y; a1[14]=(_Float16)q7.z; a1[15]=(_Float16)q7.w;
#endif

    // ---- Loop over the 8 N-tiles of the H=128 output ----
    // B layout (32x16 f16): lane<16 -> col N=lane,    VGPR v holds K=2v,2v+1 (K=0..15);
    //                       lane>=16 -> col N=lane-16, K=16+2v,17+2v     (K=16..31).
    // With W1T column-major, each fragment is 16 contiguous halves (32 bytes).
    const int colIn = lane & 15;
    const int ki    = (lane < 16) ? 0 : 1;           // kb/16 (kb = 0 or 16)
    const bool fullTile = (mtile * 16 + 16) <= N;
    const int rbase = mtile * 16 + ((lane < 16) ? 0 : 8);

    for (int nt = 0; nt < 8; ++nt) {
        const int col = nt * 16 + colIn;
        const v16h* colp = reinterpret_cast<const v16h*>(W1T + (size_t)col * H_DIM);
        v16h bt0 = colp[ki];         // W1_top rows kb..kb+15        (K half 0)
        v16h bt1 = colp[2 + ki];     // W1_top rows 32+kb..32+kb+15  (K half 1)
        v16h bb0 = colp[4 + ki];     // W1_bot rows kb..              (K half 0)
        v16h bb1 = colp[6 + ki];     // W1_bot rows 32+kb..           (K half 1)

        v8f cA = {};
        cA = __builtin_amdgcn_wmma_f32_16x16x32_f16(false, a0, false, bt0, (short)0, cA, false, false);
        cA = __builtin_amdgcn_wmma_f32_16x16x32_f16(false, a1, false, bt1, (short)0, cA, false, false);
        v8f cP = {};
        cP = __builtin_amdgcn_wmma_f32_16x16x32_f16(false, a0, false, bb0, (short)0, cP, false, false);
        cP = __builtin_amdgcn_wmma_f32_16x16x32_f16(false, a1, false, bb1, (short)0, cP, false, false);

        const float bias = b1[col];
        if (fullTile) {
            #pragma unroll
            for (int v = 0; v < 8; ++v) {
                const size_t idx = (size_t)(rbase + v) * H_DIM + col;
                Ah[idx] = (_Float16)(cA[v] + bias);
                Ph[idx] = (_Float16)(cP[v]);
            }
        } else {  // cold tail path (never taken when N % 16 == 0)
            #pragma unroll
            for (int v = 0; v < 8; ++v) {
                const int r = rbase + v;
                if (r < N) {
                    Ah[(size_t)r * H_DIM + col] = (_Float16)(cA[v] + bias);
                    Ph[(size_t)r * H_DIM + col] = (_Float16)(cP[v]);
                }
            }
        }
    }
}

// ---------------------------------------------------------------------------
// Kernel 2: per edge e (one wave32 per edge):
//   score = leaky_relu(A[dst] + P[src]) . W2 + b2
//   out[dst, e%DEG] = score; padding columns get -1e10.
// ---------------------------------------------------------------------------
__global__ __launch_bounds__(256)
void attack_edge_score(const _Float16* __restrict__ Ah,
                       const _Float16* __restrict__ Ph,
                       const float* __restrict__ W2,
                       const float* __restrict__ b2,
                       const int* __restrict__ srcIdx,
                       float* __restrict__ out,
                       int E, int DEG, int K) {
    const int gtid = blockIdx.x * blockDim.x + threadIdx.x;
    const int e    = gtid >> 5;
    const int lane = gtid & 31;
    if (e >= E) return;

    const int n    = e / DEG;        // dst node (edges sorted by dst, uniform degree)
    const int slot = e - n * DEG;
    const int s    = srcIdx[e];

    // Each lane handles 4 of the 128 hidden dims.
    const v4h* A4 = reinterpret_cast<const v4h*>(Ah);
    const v4h* P4 = reinterpret_cast<const v4h*>(Ph);
    const float4* W4 = reinterpret_cast<const float4*>(W2);

    v4h av = A4[(size_t)n * 32 + lane];
    v4h pv = P4[(size_t)s * 32 + lane];
    float4 wv = W4[lane];

    float x0 = (float)av[0] + (float)pv[0];
    float x1 = (float)av[1] + (float)pv[1];
    float x2 = (float)av[2] + (float)pv[2];
    float x3 = (float)av[3] + (float)pv[3];
    x0 = (x0 > 0.f) ? x0 : NEG_SLOPE * x0;
    x1 = (x1 > 0.f) ? x1 : NEG_SLOPE * x1;
    x2 = (x2 > 0.f) ? x2 : NEG_SLOPE * x2;
    x3 = (x3 > 0.f) ? x3 : NEG_SLOPE * x3;

    float sum = x0 * wv.x + x1 * wv.y + x2 * wv.z + x3 * wv.w;

    // wave32 tree reduction
    #pragma unroll
    for (int off = 16; off >= 1; off >>= 1)
        sum += __shfl_xor(sum, off, 32);

    if (lane == 0) {
        float* orow = out + (size_t)n * K;
        orow[slot] = sum + b2[0];
        // fill padding columns (slot+DEG, slot+2*DEG, ...) — each written exactly once
        for (int c = DEG + slot; c < K; c += DEG)
            orow[c] = -VERY_LARGE_NUMBER;
    }
}

// ---------------------------------------------------------------------------
// Host-side launcher
// Inputs: 0 node_feature [N,64] f32, 1 W1 [128,128] f32, 2 b1 [128] f32,
//         3 W2 [128,1] f32, 4 b2 [1] f32, 5 src [E] i32, 6 dst [E] i32,
//         7 maximum_num_enemy (scalar)
// Output: [N, K] f32
// ---------------------------------------------------------------------------
extern "C" void kernel_launch(void* const* d_in, const int* in_sizes, int n_in,
                              void* d_out, int out_size, void* d_ws, size_t ws_size,
                              hipStream_t stream) {
    const float* feat = (const float*)d_in[0];
    const float* W1   = (const float*)d_in[1];
    const float* b1   = (const float*)d_in[2];
    const float* W2   = (const float*)d_in[3];
    const float* b2   = (const float*)d_in[4];
    const int*   src  = (const int*)d_in[5];
    float* out = (float*)d_out;

    const int N   = in_sizes[0] / D_FEAT;     // 100000
    const int E   = in_sizes[5];              // 1.6M
    const int K   = out_size / N;             // 32
    const int DEG = E / N;                    // 16

    // Workspace layout: A (N x 128 f16), P (N x 128 f16), W1T (128x128 f16).
    _Float16* Ah  = (_Float16*)d_ws;
    _Float16* Ph  = Ah + (size_t)N * H_DIM;
    _Float16* W1T = Ph + (size_t)N * H_DIM;

    // 0) transpose+convert W1 once (32 KB, stays hot in cache)
    w1_transpose_f16<<<(2 * D_FEAT * H_DIM + 255) / 256, 256, 0, stream>>>(W1, W1T);

    // 1) WMMA precompute of A and P
    const int numMTiles = (N + 15) / 16;
    const int grid1 = (numMTiles + 7) / 8;    // 8 waves (M-tiles) per block
    attack_precompute_wmma<<<grid1, 256, 0, stream>>>(feat, W1T, b1, Ah, Ph, N, numMTiles);

    // 2) edge scoring (gather-bound, L2-resident)
    const int wavesPerBlock = 8;              // 256 threads
    const int grid2 = (E + wavesPerBlock - 1) / wavesPerBlock;
    attack_edge_score<<<grid2, 256, 0, stream>>>(Ah, Ph, W2, b2, src, out, E, DEG, K);
}
